// processor_85203561218756
// MI455X (gfx1250) — matrix operand order repaired and checked by reading, not compiled, not run on hardware
//
#include <hip/hip_runtime.h>
#include <hip/hip_bf16.h>
#include <math.h>
#include <stdint.h>

typedef __attribute__((ext_vector_type(16))) _Float16 v16h;
typedef __attribute__((ext_vector_type(8)))  float    v8f;

#define DEV __device__ __forceinline__

constexpr int B_ = 32, N_ = 1024, M_ = 64, R_ = 4, H_ = 512, NIN_ = 256, OUT_ = 256;
constexpr int P_ = 471, PP_ = 480, G_ = 2048;           // gates width = 4H
constexpr float EPS_ = 1e-8f;

// pvec slice offsets (cumsum of SIZES)
constexpr int O_WV = 0, O_EV = 64, O_FREE = 128, O_AG = 132, O_WG = 133,
              O_RM = 134, O_WKEY = 146, O_WSTR = 210, O_RKEY = 211, O_RSTR = 467;

// ---- workspace layout (float offsets) ----
constexpr size_t WS_GATES = 0;
constexpr size_t WS_H     = WS_GATES + (size_t)B_ * G_;
constexpr size_t WS_PVEC  = WS_H     + (size_t)B_ * H_;
constexpr size_t WS_FG    = WS_PVEC  + (size_t)B_ * PP_;
constexpr size_t WS_RM    = WS_FG    + (size_t)B_ * R_;
constexpr size_t WS_RSTR  = WS_RM    + (size_t)B_ * R_ * 3;
constexpr size_t WS_USAGE = WS_RSTR  + (size_t)B_ * R_;
constexpr size_t WS_ALLOC = WS_USAGE + (size_t)B_ * N_;
constexpr size_t WS_WW    = WS_ALLOC + (size_t)B_ * N_;
constexpr size_t WS_MEM   = WS_WW    + (size_t)B_ * N_;
constexpr size_t WS_RC    = WS_MEM   + (size_t)B_ * N_ * M_;
constexpr size_t WS_U16   = WS_RC    + (size_t)B_ * R_ * N_;          // f16! B*16*N halves
constexpr size_t WS_Z     = WS_U16   + (size_t)B_ * N_ * 16 / 2;
constexpr size_t WS_ZT    = WS_Z     + (size_t)B_ * N_ * 16;
constexpr size_t WS_DIAG  = WS_ZT    + (size_t)B_ * N_ * 16;
constexpr size_t WS_S     = WS_DIAG  + (size_t)B_ * N_;
constexpr size_t WS_T     = WS_S     + (size_t)B_ * R_;
constexpr size_t WS_RW    = WS_T     + (size_t)B_ * R_;
constexpr size_t WS_READS = WS_RW    + (size_t)B_ * R_ * N_;

DEV float sigm(float x)     { return 1.f / (1.f + __expf(-x)); }
DEV float softplus(float x) { return x > 20.f ? x : log1pf(__expf(x)); }

DEV float block_reduce_sum(float v, float* red) {
  int t = threadIdx.x;
  red[t] = v; __syncthreads();
  for (int s = blockDim.x >> 1; s > 0; s >>= 1) {
    if (t < s) red[t] += red[t + s];
    __syncthreads();
  }
  float r = red[0]; __syncthreads();
  return r;
}
DEV float block_reduce_max(float v, float* red) {
  int t = threadIdx.x;
  red[t] = v; __syncthreads();
  for (int s = blockDim.x >> 1; s > 0; s >>= 1) {
    if (t < s) red[t] = fmaxf(red[t], red[t + s]);
    __syncthreads();
  }
  float r = red[0]; __syncthreads();
  return r;
}

// ---------------------------------------------------------------
// WMMA fragment helpers (A/B: contiguous k order used by this file, not ISA table order; C/D as ISA)
// A (16x32 f16): lane<16 -> M=lane, kb=0 ; lane>=16 -> M=lane-16, kb=16
//                element e -> K = kb + e                   (one run of 16)
// B (32x16 f16): N = lane&15, K = e + 16*(lane>>4)         (one run of 16)
// C/D          : N = lane&15, M = g + 8*(lane>>4)
// ---------------------------------------------------------------
DEV v16h a_from_row(const float* __restrict__ row, int kA) {
  const float4 q0 = *(const float4*)(row + kA);
  const float4 q1 = *(const float4*)(row + kA + 4);
  const float4 q2 = *(const float4*)(row + kA + 8);
  const float4 q3 = *(const float4*)(row + kA + 12);
  v16h a;
  a[0]  = (_Float16)q0.x; a[1]  = (_Float16)q0.y; a[2]  = (_Float16)q0.z; a[3]  = (_Float16)q0.w;
  a[4]  = (_Float16)q1.x; a[5]  = (_Float16)q1.y; a[6]  = (_Float16)q1.z; a[7]  = (_Float16)q1.w;
  a[8]  = (_Float16)q2.x; a[9]  = (_Float16)q2.y; a[10] = (_Float16)q2.z; a[11] = (_Float16)q2.w;
  a[12] = (_Float16)q3.x; a[13] = (_Float16)q3.y; a[14] = (_Float16)q3.z; a[15] = (_Float16)q3.w;
  return a;
}

// K-segment accumulate: A rows contiguous in memory (vector loads), B gathered.
// fb(k_absolute, n) must be branch-light within a segment.
template <typename FB>
DEV v8f gemm_seg_rowA(v8f acc, const float* __restrict__ arow, FB fb,
                      int kg0, int klen, int kbA, int kbB, int nn) {
  for (int k0 = 0; k0 < klen; k0 += 32) {
    v16h a = a_from_row(arow, k0 + kbA);
    v16h b;
#pragma unroll
    for (int e = 0; e < 16; ++e) b[e] = (_Float16)fb(kg0 + k0 + kbB + e, nn);
    acc = __builtin_amdgcn_wmma_f32_16x16x32_f16(false, a, false, b,
                                                 (short)0, acc, false, false);
  }
  return acc;
}

template <typename FC>
DEV void wmma_store(const v8f& acc, FC fc, int m0, int n0, int lane) {
  int nn = n0 + (lane & 15);
  int mb = m0 + ((lane >> 4) << 3);
#pragma unroll
  for (int g = 0; g < 8; ++g) fc(mb + g, nn, acc[g]);
}

// Async-stage one batch's U^T (16 x 1024 f16 = 32KB) into LDS via ASYNCcnt path.
DEV void stage_U_lds(_Float16* Ul, const _Float16* __restrict__ src) {
  for (int i = threadIdx.x; i < 2048; i += 256) {             // 2048 x 16B chunks
    unsigned lds = (unsigned)(uintptr_t)(&Ul[i * 8]);
    const void* g = (const void*)((const char*)src + (size_t)i * 16);
    asm volatile("global_load_async_to_lds_b128 %0, %1, off"
                 :: "v"(lds), "v"(g) : "memory");
  }
  asm volatile("s_wait_asynccnt 0x0" ::: "memory");
  __syncthreads();
}

// ============ 1) LSTM gates: [X|prev_read|h0](32x1024) @ [Wx;Wh](1024x2048) ============
__global__ void __launch_bounds__(256) k_gates(
    const float* __restrict__ X, const float* __restrict__ prd,
    const float* __restrict__ h0, const float* __restrict__ Wx,
    const float* __restrict__ Wh, const float* __restrict__ bias,
    float* __restrict__ gates) {
  int wv = blockIdx.x * 8 + (threadIdx.x >> 5);
  if (wv >= 2 * (G_ / 16)) return;
  int lane = threadIdx.x & 31;
  int m0 = (wv & 1) * 16, n0 = (wv >> 1) * 16;
  int mrow = lane & 15, kbA = (lane < 16) ? 0 : 16, kbB = (lane < 16) ? 0 : 16;
  int nn = n0 + (lane & 15);
  auto fbx = [=](int k, int n) -> float { return Wx[k * G_ + n]; };
  auto fbh = [=](int k, int n) -> float { return Wh[(k - 512) * G_ + n]; };
  v8f acc = {};
  acc = gemm_seg_rowA(acc, X   + (size_t)(m0 + mrow) * NIN_, fbx, 0,   256, kbA, kbB, nn);
  acc = gemm_seg_rowA(acc, prd + (size_t)(m0 + mrow) * NIN_, fbx, 256, 256, kbA, kbB, nn);
  acc = gemm_seg_rowA(acc, h0  + (size_t)(m0 + mrow) * H_,   fbh, 512, 512, kbA, kbB, nn);
  wmma_store(acc, [=](int m, int n, float v) { gates[m * G_ + n] = v + bias[n]; },
             m0, n0, lane);
}

// ============ 2) LSTM activations -> h ============
__global__ void k_lstm_act(const float* __restrict__ gates, const float* __restrict__ c0,
                           float* __restrict__ h) {
  int id = blockIdx.x * blockDim.x + threadIdx.x;
  if (id >= B_ * H_) return;
  int b = id / H_, j = id % H_;
  const float* g = gates + (size_t)b * G_;
  float iv = sigm(g[j]), fv = sigm(g[j + H_]);
  float gv = tanhf(g[j + 2 * H_]), ov = sigm(g[j + 3 * H_]);
  float c  = fv * c0[id] + iv * gv;
  h[id] = ov * tanhf(c);
}

// ============ 3) para projection: [h|emb](32x1024) @ para_W(1024x471) ============
__global__ void __launch_bounds__(256) k_para(
    const float* __restrict__ h, const float* __restrict__ emb,
    const float* __restrict__ Wp, const float* __restrict__ bp,
    float* __restrict__ pvec) {
  int wv = blockIdx.x * 8 + (threadIdx.x >> 5);
  if (wv >= 2 * (PP_ / 16)) return;
  int lane = threadIdx.x & 31;
  int m0 = (wv & 1) * 16, n0 = (wv >> 1) * 16;
  int mrow = lane & 15, kbA = (lane < 16) ? 0 : 16, kbB = (lane < 16) ? 0 : 16;
  int nn = n0 + (lane & 15);
  auto fbp = [=](int k, int n) -> float { return (n < P_) ? Wp[k * P_ + n] : 0.f; };
  v8f acc = {};
  acc = gemm_seg_rowA(acc, h   + (size_t)(m0 + mrow) * H_, fbp, 0,   512, kbA, kbB, nn);
  acc = gemm_seg_rowA(acc, emb + (size_t)(m0 + mrow) * H_, fbp, 512, 512, kbA, kbB, nn);
  wmma_store(acc, [=](int m, int n, float v) {
    if (n < P_) pvec[m * PP_ + n] = v + bp[n];
  }, m0, n0, lane);
}

// ============ 4) layernorm over 471 elems per row ============
__global__ void k_layernorm(float* __restrict__ pvec, const float* __restrict__ lg,
                            const float* __restrict__ lb) {
  __shared__ float r1[512];
  __shared__ float r2[512];
  int b = blockIdx.x, t = threadIdx.x;
  float x = (t < P_) ? pvec[b * PP_ + t] : 0.f;
  r1[t] = x; r2[t] = x * x; __syncthreads();
  for (int s = 256; s > 0; s >>= 1) {
    if (t < s) { r1[t] += r1[t + s]; r2[t] += r2[t + s]; }
    __syncthreads();
  }
  float mu  = r1[0] / P_;
  float var = r2[0] / P_ - mu * mu;
  float inv = rsqrtf(var + 1e-5f);
  if (t < P_) pvec[b * PP_ + t] = (x - mu) * inv * lg[t] + lb[t];
}

// ============ 5) per-batch small params ============
__global__ void k_small(const float* __restrict__ pvec, float* __restrict__ fg,
                        float* __restrict__ rm, float* __restrict__ rstr) {
  int b = blockIdx.x, t = threadIdx.x;
  const float* pv = pvec + (size_t)b * PP_;
  if (t < R_) {
    fg[b * R_ + t] = sigm(pv[O_FREE + t]);
    float a0 = pv[O_RM + t * 3], a1 = pv[O_RM + t * 3 + 1], a2 = pv[O_RM + t * 3 + 2];
    float mx = fmaxf(a0, fmaxf(a1, a2));
    float e0 = __expf(a0 - mx), e1 = __expf(a1 - mx), e2 = __expf(a2 - mx);
    float s  = e0 + e1 + e2;
    rm[b * 12 + t * 3 + 0] = e0 / s;
    rm[b * 12 + t * 3 + 1] = e1 / s;
    rm[b * 12 + t * 3 + 2] = e2 / s;
    rstr[b * R_ + t] = 1.f + softplus(pv[O_RSTR + t]);
  }
}

// ============ 6) usage (with retention psi) ============
__global__ void k_usage(const float* __restrict__ fg, const float* __restrict__ rwp,
                        const float* __restrict__ usage_prev, const float* __restrict__ wwp,
                        float* __restrict__ usage) {
  int id = blockIdx.x * blockDim.x + threadIdx.x;
  if (id >= B_ * N_) return;
  int b = id >> 10, n = id & (N_ - 1);
  float psi = 1.f;
#pragma unroll
  for (int r = 0; r < R_; ++r)
    psi *= 1.f - fg[b * R_ + r] * rwp[((size_t)b * R_ + r) * N_ + n];
  float up = usage_prev[id], wp = wwp[id];
  usage[id] = (up + wp - up * wp) * psi;
}

// ============ 7) per-batch bitonic argsort + allocation weights ============
__global__ void __launch_bounds__(1024) k_sort_alloc(const float* __restrict__ usage,
                                                     float* __restrict__ alloc) {
  __shared__ float key[N_];
  __shared__ int   idx[N_];
  __shared__ float cp[N_];
  int b = blockIdx.x, t = threadIdx.x;
  key[t] = usage[b * N_ + t];
  idx[t] = t;
  __syncthreads();
  for (int ksz = 2; ksz <= N_; ksz <<= 1) {
    for (int j = ksz >> 1; j > 0; j >>= 1) {
      int ixj = t ^ j;
      if (ixj > t) {
        bool up = ((t & ksz) == 0);
        float a = key[t], c = key[ixj];
        if (up ? (a > c) : (a < c)) {
          key[t] = c; key[ixj] = a;
          int ti = idx[t]; idx[t] = idx[ixj]; idx[ixj] = ti;
        }
      }
      __syncthreads();
    }
  }
  cp[t] = key[t]; __syncthreads();
  for (int off = 1; off < N_; off <<= 1) {
    float v = cp[t];
    float u = (t >= off) ? cp[t - off] : 1.f;
    __syncthreads();
    cp[t] = v * u;
    __syncthreads();
  }
  alloc[b * N_ + idx[t]] = (1.f - key[t]) * cp[t];
}

// ============ 8) write content weighting + final write weights ww ============
__global__ void k_wc_ww(const float* __restrict__ pvec, const float* __restrict__ memory,
                        const float* __restrict__ alloc, float* __restrict__ ww) {
  __shared__ float sc[N_];
  __shared__ float red[256];
  int b = blockIdx.x, t = threadIdx.x;
  const float* pv = pvec + (size_t)b * PP_;
  float kn2 = 0.f;
#pragma unroll
  for (int m = 0; m < M_; ++m) { float kv = pv[O_WKEY + m]; kn2 += kv * kv; }
  float knorm = sqrtf(kn2) + EPS_;
  float beta  = 1.f + softplus(pv[O_WSTR]);
  for (int n = t; n < N_; n += 256) {
    const float* mr = memory + ((size_t)b * N_ + n) * M_;
    float dot = 0.f, mn2 = 0.f;
#pragma unroll
    for (int m = 0; m < M_; ++m) {
      float mv = mr[m];
      dot += pv[O_WKEY + m] * mv;
      mn2 += mv * mv;
    }
    sc[n] = beta * dot / (knorm * (sqrtf(mn2) + EPS_));
  }
  __syncthreads();
  float mx = -3.4e38f;
  for (int n = t; n < N_; n += 256) mx = fmaxf(mx, sc[n]);
  mx = block_reduce_max(mx, red);
  float se = 0.f;
  for (int n = t; n < N_; n += 256) { float e = __expf(sc[n] - mx); sc[n] = e; se += e; }
  se = block_reduce_sum(se, red);
  float ag = sigm(pv[O_AG]), wg = sigm(pv[O_WG]);
  for (int n = t; n < N_; n += 256) {
    float wc = sc[n] / se;
    ww[b * N_ + n] = wg * (ag * alloc[b * N_ + n] + (1.f - ag) * wc);
  }
}

// ============ 9) memory write (erase + add), W=1 ============
__global__ void k_mem(const float* __restrict__ memory, const float* __restrict__ ww,
                      const float* __restrict__ pvec, float* __restrict__ mem_new) {
  int id = blockIdx.x * blockDim.x + threadIdx.x;
  if (id >= B_ * N_ * M_) return;
  int b = id >> 16, rem = id & 65535;
  int n = rem >> 6, m = rem & 63;
  const float* pv = pvec + (size_t)b * PP_;
  float w = ww[b * N_ + n];
  float e = sigm(pv[O_EV + m]);
  float v = pv[O_WV + m];
  mem_new[id] = memory[id] * (1.f - w * e) + w * v;
}

// ============ 10) read content weighting on updated memory ============
__global__ void k_rc(const float* __restrict__ pvec, const float* __restrict__ mem_new,
                     const float* __restrict__ rstr, float* __restrict__ rc) {
  __shared__ float sc[N_];
  __shared__ float red[256];
  int b = blockIdx.x >> 2, r = blockIdx.x & 3, t = threadIdx.x;
  const float* pv  = pvec + (size_t)b * PP_ + O_RKEY + r * M_;
  const float* mem = mem_new + (size_t)b * N_ * M_;
  float kn2 = 0.f;
#pragma unroll
  for (int m = 0; m < M_; ++m) { float kv = pv[m]; kn2 += kv * kv; }
  float knorm = sqrtf(kn2) + EPS_;
  float beta  = rstr[b * R_ + r];
  for (int n = t; n < N_; n += 256) {
    const float* mr = mem + (size_t)n * M_;
    float dot = 0.f, mn2 = 0.f;
#pragma unroll
    for (int m = 0; m < M_; ++m) { float mv = mr[m]; dot += pv[m] * mv; mn2 += mv * mv; }
    sc[n] = beta * dot / (knorm * (sqrtf(mn2) + EPS_));
  }
  __syncthreads();
  float mx = -3.4e38f;
  for (int n = t; n < N_; n += 256) mx = fmaxf(mx, sc[n]);
  mx = block_reduce_max(mx, red);
  float se = 0.f;
  for (int n = t; n < N_; n += 256) { float e = __expf(sc[n] - mx); sc[n] = e; se += e; }
  se = block_reduce_sum(se, red);
  for (int n = t; n < N_; n += 256)
    rc[((size_t)b * R_ + r) * N_ + n] = sc[n] / se;
}

// ============ 11) build U^T (f16, 16 x N per batch) + diag(L) ============
__global__ void k_prepU(const float* __restrict__ rwp, const float* __restrict__ ww,
                        const float* __restrict__ Lp, _Float16* __restrict__ U16,
                        float* __restrict__ diag) {
  int id = blockIdx.x * blockDim.x + threadIdx.x;
  if (id >= B_ * N_) return;
  int b = id >> 10, n = id & (N_ - 1);
  float w = ww[id];
  _Float16* u = U16 + (size_t)b * 16 * N_;
#pragma unroll
  for (int r = 0; r < R_; ++r) {
    float x = rwp[((size_t)b * R_ + r) * N_ + n];
    u[r * N_ + n]       = (_Float16)x;
    u[(4 + r) * N_ + n] = (_Float16)(w * x);
  }
#pragma unroll
  for (int c = 8; c < 16; ++c) u[c * N_ + n] = (_Float16)0.f;
  diag[id] = Lp[(size_t)b * N_ * N_ + (size_t)n * N_ + n];
}

// ============ 12) scalar reductions S_r = p.rwp_r, T_r = w.rwp_r ============
__global__ void k_ST(const float* __restrict__ prec, const float* __restrict__ ww,
                     const float* __restrict__ rwp, float* __restrict__ S,
                     float* __restrict__ T) {
  __shared__ float red[256];
  int b = blockIdx.x, t = threadIdx.x;
  float s[R_] = {}, tt[R_] = {};
  for (int n = t; n < N_; n += 256) {
    float p = prec[b * N_ + n], w = ww[b * N_ + n];
#pragma unroll
    for (int r = 0; r < R_; ++r) {
      float x = rwp[((size_t)b * R_ + r) * N_ + n];
      s[r]  += p * x;
      tt[r] += w * x;
    }
  }
#pragma unroll
  for (int r = 0; r < R_; ++r) {
    float v = block_reduce_sum(s[r], red);
    if (t == 0) S[b * R_ + r] = v;
    v = block_reduce_sum(tt[r], red);
    if (t == 0) T[b * R_ + r] = v;
  }
}

// ============ 13) Z = L @ U : HBM-bound pass, vector A loads, LDS-staged U ============
__global__ void __launch_bounds__(256) k_Z(const float* __restrict__ Lp,
                                           const _Float16* __restrict__ U16,
                                           float* __restrict__ Z) {
  __shared__ _Float16 Ul[16 * N_];                  // 32KB staged U^T
  int b = blockIdx.x >> 3, grp = blockIdx.x & 7;
  stage_U_lds(Ul, U16 + (size_t)b * 16 * N_);
  int wavei = threadIdx.x >> 5, lane = threadIdx.x & 31;
  int m0 = (grp * 8 + wavei) * 16;
  int mrow = lane & 15, kbA = (lane < 16) ? 0 : 16, kbB = (lane < 16) ? 0 : 16;
  int nn = lane & 15;
  const float* row = Lp + (size_t)b * N_ * N_ + (size_t)(m0 + mrow) * N_;
  const _Float16* bl = Ul + (size_t)nn * N_;
  v8f acc = {};
  for (int k0 = 0; k0 < N_; k0 += 32) {
    if (k0 + 256 < N_) __builtin_prefetch(row + k0 + 256, 0, 1);
    v16h a = a_from_row(row, k0 + kbA);
    v16h bv = *(const v16h*)(bl + k0 + kbB);        // contiguous 32B LDS read
    acc = __builtin_amdgcn_wmma_f32_16x16x32_f16(false, a, false, bv,
                                                 (short)0, acc, false, false);
  }
  wmma_store(acc, [=](int m, int n, float v) {
    Z[((size_t)b * N_ + m) * 16 + n] = v;
  }, m0, 0, lane);
}

// ============ 14) Zt = L^T @ U : second pass hits 192MB L2 ============
__global__ void __launch_bounds__(256) k_Zt(const float* __restrict__ Lp,
                                            const _Float16* __restrict__ U16,
                                            float* __restrict__ ZT) {
  __shared__ _Float16 Ul[16 * N_];
  int b = blockIdx.x >> 3, grp = blockIdx.x & 7;
  stage_U_lds(Ul, U16 + (size_t)b * 16 * N_);
  int wavei = threadIdx.x >> 5, lane = threadIdx.x & 31;
  int m0 = (grp * 8 + wavei) * 16;
  int mrow = lane & 15, kbA = (lane < 16) ? 0 : 16, kbB = (lane < 16) ? 0 : 16;
  int nn = lane & 15;
  const float* lp = Lp + (size_t)b * N_ * N_ + (m0 + mrow);  // column m0+mrow
  const _Float16* bl = Ul + (size_t)nn * N_;
  v8f acc = {};
  for (int k0 = 0; k0 < N_; k0 += 32) {
    v16h a;
#pragma unroll
    for (int e = 0; e < 16; ++e) {
      int k = k0 + kbA + e;
      a[e] = (_Float16)lp[(size_t)k * N_];          // coalesced across lanes
    }
    v16h bv = *(const v16h*)(bl + k0 + kbB);
    acc = __builtin_amdgcn_wmma_f32_16x16x32_f16(false, a, false, bv,
                                                 (short)0, acc, false, false);
  }
  wmma_store(acc, [=](int m, int n, float v) {
    ZT[((size_t)b * N_ + m) * 16 + n] = v;
  }, m0, 0, lane);
}

// ============ 15) assemble fwd/bwd analytically, combine into read weights ============
__global__ void k_rw(const float* __restrict__ ww, const float* __restrict__ prec,
                     const float* __restrict__ diag, const float* __restrict__ rwp,
                     const float* __restrict__ Z, const float* __restrict__ ZT,
                     const float* __restrict__ S, const float* __restrict__ T,
                     const float* __restrict__ rm, const float* __restrict__ rc,
                     float* __restrict__ rw) {
  int id = blockIdx.x * blockDim.x + threadIdx.x;
  if (id >= B_ * N_) return;
  int b = id >> 10, i = id & (N_ - 1);
  float w = ww[id], p = prec[b * N_ + i], d = diag[id];
  const float* z  = Z  + (size_t)id * 16;
  const float* zt = ZT + (size_t)id * 16;
#pragma unroll
  for (int r = 0; r < R_; ++r) {
    float x    = rwp[((size_t)b * R_ + r) * N_ + i];
    float corr = ((1.f - 2.f * w) * d + w * p) * x;   // remove i==j diagonal term
    float fwd  = (1.f - w) * z[r]  - z[4 + r]  + w * S[b * R_ + r] - corr;
    float bwd  = (1.f - w) * zt[r] - zt[4 + r] + p * T[b * R_ + r] - corr;
    const float* m = rm + b * 12 + r * 3;
    rw[((size_t)b * R_ + r) * N_ + i] =
        m[1] * rc[((size_t)b * R_ + r) * N_ + i] + m[0] * bwd + m[2] * fwd;
  }
}

// ============ 16) reads = rw(4x1024, padded to 16) @ mem_new(1024x64) ============
__global__ void __launch_bounds__(256) k_reads(const float* __restrict__ rw,
                                               const float* __restrict__ mem_new,
                                               float* __restrict__ reads) {
  int wv = blockIdx.x * 8 + (threadIdx.x >> 5);
  if (wv >= B_ * (M_ / 16)) return;
  int lane = threadIdx.x & 31;
  int b = wv >> 2, n0 = (wv & 3) * 16;
  int mrow = lane & 15, kbA = (lane < 16) ? 0 : 16, kbB = (lane < 16) ? 0 : 16;
  int nn = n0 + (lane & 15);
  const float* ar = rw + (size_t)b * R_ * N_ + (size_t)mrow * N_;
  const float* mb = mem_new + (size_t)b * N_ * M_;
  v8f acc = {};
  for (int k0 = 0; k0 < N_; k0 += 32) {
    v16h a = {};
    if (mrow < R_) a = a_from_row(ar, k0 + kbA);
    v16h bv;
#pragma unroll
    for (int e = 0; e < 16; ++e)
      bv[e] = (_Float16)mb[(size_t)(k0 + kbB + e) * M_ + nn];
    acc = __builtin_amdgcn_wmma_f32_16x16x32_f16(false, a, false, bv,
                                                 (short)0, acc, false, false);
  }
  wmma_store(acc, [=](int m, int n, float v) {
    if (m < R_) reads[(size_t)b * R_ * M_ + m * M_ + n] = v;
  }, 0, n0, lane);
}

// ============ 17) out = [h|emb|reads](32x1280) @ proc_W(1280x256) + b ============
__global__ void __launch_bounds__(256) k_out(
    const float* __restrict__ h, const float* __restrict__ emb,
    const float* __restrict__ reads, const float* __restrict__ Wo,
    const float* __restrict__ bo, float* __restrict__ out) {
  int wv = blockIdx.x * 8 + (threadIdx.x >> 5);
  if (wv >= 2 * (OUT_ / 16)) return;
  int lane = threadIdx.x & 31;
  int m0 = (wv & 1) * 16, n0 = (wv >> 1) * 16;
  int mrow = lane & 15, kbA = (lane < 16) ? 0 : 16, kbB = (lane < 16) ? 0 : 16;
  int nn = n0 + (lane & 15);
  auto fbo = [=](int k, int n) -> float { return Wo[k * OUT_ + n]; };
  v8f acc = {};
  acc = gemm_seg_rowA(acc, h     + (size_t)(m0 + mrow) * H_,        fbo, 0,    512, kbA, kbB, nn);
  acc = gemm_seg_rowA(acc, emb   + (size_t)(m0 + mrow) * H_,        fbo, 512,  512, kbA, kbB, nn);
  acc = gemm_seg_rowA(acc, reads + (size_t)(m0 + mrow) * (R_ * M_), fbo, 1024, 256, kbA, kbB, nn);
  wmma_store(acc, [=](int m, int n, float v) { out[m * OUT_ + n] = v + bo[n]; },
             m0, n0, lane);
}

// ==================================================================
extern "C" void kernel_launch(void* const* d_in, const int* in_sizes, int n_in,
                              void* d_out, int out_size, void* d_ws, size_t ws_size,
                              hipStream_t stream) {
  const float* X    = (const float*)d_in[0];
  const float* emb  = (const float*)d_in[1];
  const float* mem0 = (const float*)d_in[2];
  const float* prd  = (const float*)d_in[3];
  const float* h0   = (const float*)d_in[4];
  const float* c0   = (const float*)d_in[5];
  const float* rwp  = (const float*)d_in[6];
  const float* wwp  = (const float*)d_in[7];
  const float* usp  = (const float*)d_in[8];
  const float* prc  = (const float*)d_in[9];
  const float* Lp   = (const float*)d_in[10];
  const float* Wx   = (const float*)d_in[11];
  const float* Wh   = (const float*)d_in[12];
  const float* lb   = (const float*)d_in[13];
  const float* Wp   = (const float*)d_in[14];
  const float* pb   = (const float*)d_in[15];
  const float* lng  = (const float*)d_in[16];
  const float* lnb  = (const float*)d_in[17];
  const float* Wo   = (const float*)d_in[18];
  const float* bo   = (const float*)d_in[19];
  float* out = (float*)d_out;
  float* ws  = (float*)d_ws;

  float*     gates = ws + WS_GATES;  float* h     = ws + WS_H;
  float*     pvec  = ws + WS_PVEC;   float* fg    = ws + WS_FG;
  float*     rm    = ws + WS_RM;     float* rstr  = ws + WS_RSTR;
  float*     usage = ws + WS_USAGE;  float* alloc = ws + WS_ALLOC;
  float*     wwv   = ws + WS_WW;     float* memn  = ws + WS_MEM;
  float*     rc    = ws + WS_RC;     _Float16* U16 = (_Float16*)(ws + WS_U16);
  float*     Z     = ws + WS_Z;      float* ZT    = ws + WS_ZT;
  float*     diag  = ws + WS_DIAG;   float* S     = ws + WS_S;
  float*     T     = ws + WS_T;      float* rw    = ws + WS_RW;
  float*     reads = ws + WS_READS;

  k_gates     <<<32,   256, 0, stream>>>(X, prd, h0, Wx, Wh, lb, gates);
  k_lstm_act  <<<64,   256, 0, stream>>>(gates, c0, h);
  k_para      <<<8,    256, 0, stream>>>(h, emb, Wp, pb, pvec);
  k_layernorm <<<32,   512, 0, stream>>>(pvec, lng, lnb);
  k_small     <<<32,    32, 0, stream>>>(pvec, fg, rm, rstr);
  k_usage     <<<128,  256, 0, stream>>>(fg, rwp, usp, wwp, usage);
  k_sort_alloc<<<32,  1024, 0, stream>>>(usage, alloc);
  k_wc_ww     <<<32,   256, 0, stream>>>(pvec, mem0, alloc, wwv);
  k_mem       <<<8192, 256, 0, stream>>>(mem0, wwv, pvec, memn);
  k_rc        <<<128,  256, 0, stream>>>(pvec, memn, rstr, rc);
  k_prepU     <<<128,  256, 0, stream>>>(rwp, wwv, Lp, U16, diag);
  k_ST        <<<32,   256, 0, stream>>>(prc, wwv, rwp, S, T);
  k_Z         <<<256,  256, 0, stream>>>(Lp, U16, Z);
  k_Zt        <<<256,  256, 0, stream>>>(Lp, U16, ZT);
  k_rw        <<<128,  256, 0, stream>>>(wwv, prc, diag, rwp, Z, ZT, S, T, rm, rc, rw);
  k_reads     <<<16,   256, 0, stream>>>(rw, memn, reads);
  k_out       <<<4,    256, 0, stream>>>(h, emb, reads, Wo, bo, out);
}
